// Multi_Scale_Attention5_17523466568050
// MI455X (gfx1250) — matrix operand order repaired and checked by reading, not compiled, not run on hardware
//
#include <hip/hip_runtime.h>

typedef __attribute__((ext_vector_type(16))) __bf16 v16bf;
typedef __attribute__((ext_vector_type(8)))  __bf16 v8bf;
typedef __attribute__((ext_vector_type(4)))  __bf16 v4bf;
typedef __attribute__((ext_vector_type(8)))  float  v8f;
typedef __attribute__((ext_vector_type(4)))  float  v4f;

union FragU { v16bf v; struct { v8bf lo; v8bf hi; } h; };

__device__ __forceinline__ v8f wmma_bf16(v16bf a, v16bf b, v8f c) {
  // D = A(16x32 bf16) x B(32x16 bf16) + C(16x16 f32)
  return __builtin_amdgcn_wmma_f32_16x16x32_bf16(false, a, false, b, (short)0, c,
                                                 false, false);
}

// A-matrix fragment from row-major bf16 [m][k] tile in LDS.
// A 16x32 layout (matches B frag): lane(0-15)=row m, elems 0..15 -> K=16g..16g+15
// (g = lane>>4).
__device__ __forceinline__ v16bf load_a_frag(const __bf16* base, int lda, int k0) {
  const int lane = threadIdx.x & 31;
  const int m = lane & 15, g = lane >> 4;
  const __bf16* p = base + m * lda + k0 + 16 * g;
  FragU f;
  f.h.lo = *(const v8bf*)(p);
  f.h.hi = *(const v8bf*)(p + 8);
  return f.v;
}

// B-matrix fragment from a K-transposed bf16 [n][k] tile in LDS.
// B layout used here (same k order as A frag): lane(0-15)=col n holds K=0..15, lanes 16-31 hold K=16..31.
__device__ __forceinline__ v16bf load_b_frag(const __bf16* baseT, int ldk, int k0) {
  const int lane = threadIdx.x & 31;
  const int n = lane & 15, g = lane >> 4;
  return *(const v16bf*)(baseT + n * ldk + k0 + 16 * g);
}

// wave32 butterfly reduction (sum) of two scalars
__device__ __forceinline__ void wave_red2(float& a, float& b) {
#pragma unroll
  for (int off = 16; off > 0; off >>= 1) {
    a += __shfl_xor(a, off, 32);
    b += __shfl_xor(b, off, 32);
  }
}

// ---------------------------------------------------------------------------
// 1x1 conv as GEMM:  Out[m,n] = sum_k W[m,k] * X[k,n] + bias[m]   (per batch)
// Block: 256 thr (8 waves). Tile M=128 (whole), N=64. Whole W kept in LDS.
// BF16IN selects the input dtype at compile time (no runtime branch).
// ---------------------------------------------------------------------------
template <bool BF16IN>
__global__ void __launch_bounds__(256)
k_gemm1x1(const float* __restrict__ Wsrc, const float* __restrict__ bias,
          const float* __restrict__ Xf, const __bf16* __restrict__ Xh,
          __bf16* __restrict__ Out, int C, int HW)
{
  __shared__ __align__(32) __bf16 Wb[128 * 128];  // [m][k]
  __shared__ __align__(32) __bf16 Xt[64 * 32];    // [n][k] transposed tile
  const int tid = threadIdx.x, wv = tid >> 5;
  const int lane = tid & 31, ln = lane & 15, g = lane >> 4;
  const int nb = blockIdx.x * 64;
  const int batch = blockIdx.y;

  for (int i = tid; i < C * C; i += 256) Wb[i] = (__bf16)Wsrc[i];

  // staging role: this thread owns column n, k-chunk kg*8..kg*8+7
  const int n = tid & 63, kg = tid >> 6;

  v8f acc[4] = {};
  const size_t xbase = (size_t)batch * C * HW + nb + n;
#pragma unroll
  for (int k0 = 0; k0 < 128; k0 += 32) {
    __syncthreads();
    {
      const size_t sb = xbase + (size_t)(k0 + kg * 8) * HW;
      v8bf pk;
#pragma unroll
      for (int j = 0; j < 8; ++j) {
        if constexpr (BF16IN) pk[j] = Xh[sb + (size_t)j * HW];
        else                  pk[j] = (__bf16)Xf[sb + (size_t)j * HW];
      }
      *(v8bf*)&Xt[n * 32 + kg * 8] = pk;   // one b128 LDS store
    }
    __syncthreads();
    v16bf a = load_a_frag(Wb + wv * 16 * C, C, k0);
#pragma unroll
    for (int nt = 0; nt < 4; ++nt) {
      v16bf b = load_b_frag(Xt + nt * 16 * 32, 32, 0);
      acc[nt] = wmma_bf16(a, b, acc[nt]);
    }
  }
#pragma unroll
  for (int v = 0; v < 8; ++v) {
    int m = wv * 16 + v + 8 * g;                  // D layout: row = v + 8g
    float bv = bias[m];
#pragma unroll
    for (int nt = 0; nt < 4; ++nt)
      Out[(size_t)(batch * C + m) * HW + nb + nt * 16 + ln] =
          (__bf16)(acc[nt][v] + bv);
  }
}

// ---------------------------------------------------------------------------
// Attention slice kernel: one workgroup per (b,h).
//   S  = inorm( X1slice(c x w) * XFPslice(d x w)^T )   -> 128x128
//   F  = inorm( S * XPslice(c x w) )                   -> 128x128
// Result written (bf16) into channel block `chBlock` of the concat buffer.
// ---------------------------------------------------------------------------
__global__ void __launch_bounds__(256)
k_attn(const __bf16* __restrict__ Xa, const __bf16* __restrict__ Xb,
       const __bf16* __restrict__ XP, __bf16* __restrict__ OutCat, int chBlock)
{
  const int C = 128, H = 128, Wd = 128;
  __shared__ __align__(32) union SMU {
    struct { __bf16 A1[128 * 128]; __bf16 B1[128 * 128]; } p1;  // phase 1
    struct { __bf16 Sb[128 * 128]; __bf16 XPt[128 * 128]; } p2; // phase 2
  } sm;
  __shared__ float red[16];

  const int tid = threadIdx.x, wv = tid >> 5;
  const int lane = tid & 31, ln = lane & 15, g = lane >> 4;
  const int h = blockIdx.x, batch = blockIdx.y;
  const int m0 = wv * 16;

  // stage X1 / XFP slices, row-major [c][w]  (B-frags of S need rows of XFP)
  for (int i = tid; i < C * Wd / 8; i += 256) {
    int c = i >> 4, w = (i & 15) * 8;
    size_t s = ((size_t)(batch * C + c) * H + h) * Wd + w;
    *(v8bf*)&sm.p1.A1[c * Wd + w] = *(const v8bf*)&Xa[s];
    *(v8bf*)&sm.p1.B1[c * Wd + w] = *(const v8bf*)&Xb[s];
  }
  __syncthreads();

  v8f accS[8] = {};
#pragma unroll
  for (int k0 = 0; k0 < Wd; k0 += 32) {
    v16bf a = load_a_frag(sm.p1.A1 + m0 * Wd, Wd, k0);
#pragma unroll
    for (int nt = 0; nt < 8; ++nt) {
      v16bf b = load_b_frag(sm.p1.B1 + nt * 16 * Wd, Wd, k0);
      accS[nt] = wmma_bf16(a, b, accS[nt]);
    }
  }
  // instance-norm stats over the whole 128x128 S (wave shfl + LDS combine)
  float s1 = 0.f, s2 = 0.f;
#pragma unroll
  for (int nt = 0; nt < 8; ++nt)
#pragma unroll
    for (int v = 0; v < 8; ++v) { float x = accS[nt][v]; s1 += x; s2 += x * x; }
  wave_red2(s1, s2);
  if (lane == 0) { red[wv] = s1; red[8 + wv] = s2; }
  __syncthreads();
  {
    float a = 0.f, b2 = 0.f;
#pragma unroll
    for (int i = 0; i < 8; ++i) { a += red[i]; b2 += red[8 + i]; }
    s1 = a * (1.f / 16384.f);                       // mean
    s2 = rsqrtf(b2 * (1.f / 16384.f) - s1 * s1 + 1e-5f);  // rstd
  }
  const float mean = s1, rstd = s2;

  // normalized S -> LDS row-major bf16 (overlays A1; all reads are done)
#pragma unroll
  for (int nt = 0; nt < 8; ++nt)
#pragma unroll
    for (int v = 0; v < 8; ++v)
      sm.p2.Sb[(m0 + v + 8 * g) * C + nt * 16 + ln] =
          (__bf16)((accS[nt][v] - mean) * rstd);

  // stage XP slice transposed [w][c] (overlays B1)
  for (int i = tid; i < C * Wd / 8; i += 256) {
    int c = i >> 4, w = (i & 15) * 8;
    size_t s = ((size_t)(batch * C + c) * H + h) * Wd + w;
    v8bf row = *(const v8bf*)&XP[s];
#pragma unroll
    for (int j = 0; j < 8; ++j) sm.p2.XPt[(w + j) * C + c] = row[j];
  }
  __syncthreads();

  v8f accF[8] = {};
#pragma unroll
  for (int k0 = 0; k0 < C; k0 += 32) {
    v16bf a = load_a_frag(sm.p2.Sb + m0 * C, C, k0);
#pragma unroll
    for (int nt = 0; nt < 8; ++nt) {
      v16bf b = load_b_frag(sm.p2.XPt + nt * 16 * C, C, k0);
      accF[nt] = wmma_bf16(a, b, accF[nt]);
    }
  }
  float t1 = 0.f, t2 = 0.f;
#pragma unroll
  for (int nt = 0; nt < 8; ++nt)
#pragma unroll
    for (int v = 0; v < 8; ++v) { float x = accF[nt][v]; t1 += x; t2 += x * x; }
  wave_red2(t1, t2);
  __syncthreads();                                  // red reuse
  if (lane == 0) { red[wv] = t1; red[8 + wv] = t2; }
  __syncthreads();
  {
    float a = 0.f, b2 = 0.f;
#pragma unroll
    for (int i = 0; i < 8; ++i) { a += red[i]; b2 += red[8 + i]; }
    t1 = a * (1.f / 16384.f);
    t2 = rsqrtf(b2 * (1.f / 16384.f) - t1 * t1 + 1e-5f);
  }
  const float mean2 = t1, rstd2 = t2;

#pragma unroll
  for (int v = 0; v < 8; ++v) {
    int m = m0 + v + 8 * g;
    size_t ob = ((size_t)(batch * 3 * C + chBlock * C + m) * H + h) * Wd;
#pragma unroll
    for (int nt = 0; nt < 8; ++nt)
      OutCat[ob + nt * 16 + ln] = (__bf16)((accF[nt][v] - mean2) * rstd2);
  }
}

// ---------------------------------------------------------------------------
// 3x3 conv (384->384) as implicit GEMM over 9 taps, fused BN + ReLU.
// Block tile: M=128, N=64 (along w). K loop = 9 taps x 12 chunks of 32.
// ---------------------------------------------------------------------------
__global__ void __launch_bounds__(256)
k_conv3x3(const __bf16* __restrict__ In, const __bf16* __restrict__ Wt,
          const float* __restrict__ bm, const float* __restrict__ gamma,
          const float* __restrict__ beta, const float* __restrict__ rmean,
          const float* __restrict__ rvar, float* __restrict__ Outp)
{
  const int Cin = 384, H = 128, Wd = 128;
  __shared__ __align__(32) __bf16 Aw[128 * 32];   // [m][k]
  __shared__ __align__(32) __bf16 Bt[64 * 32];    // [n][k]
  const int tid = threadIdx.x, wv = tid >> 5;
  const int lane = tid & 31, ln = lane & 15, g = lane >> 4;
  const int h = blockIdx.x >> 1, w0 = (blockIdx.x & 1) * 64;
  const int mb = blockIdx.y, batch = blockIdx.z;
  const int n = tid & 63, kg = tid >> 6;          // B-staging role

  v8f acc[4] = {};
  for (int tap = 0; tap < 9; ++tap) {
    const int dh = tap / 3 - 1, dw = tap % 3 - 1;
    const int hh = h + dh;
    const bool rok = (unsigned)hh < (unsigned)H;
    const int ww = w0 + n + dw;
    const bool ok = rok && (unsigned)ww < (unsigned)Wd;
    const __bf16* wtap = Wt + (size_t)tap * Cin * Cin + (size_t)mb * 128 * Cin;
    for (int kc = 0; kc < Cin; kc += 32) {
      __syncthreads();
#pragma unroll
      for (int j = 0; j < 2; ++j) {               // 128x32 weight tile, b128 ops
        int i8 = tid + j * 256;                   // covers 8 elems each
        int mi = i8 >> 2, k8 = (i8 & 3) * 8;
        *(v8bf*)&Aw[mi * 32 + k8] =
            *(const v8bf*)&wtap[(size_t)mi * Cin + kc + k8];
      }
      {                                           // 64x32 input tile, padded
        v8bf pk;
#pragma unroll
        for (int j = 0; j < 8; ++j) pk[j] = (__bf16)0.f;
        if (ok) {
          const __bf16* src =
              In + ((size_t)(batch * Cin + kc + kg * 8) * H + hh) * Wd + ww;
#pragma unroll
          for (int j = 0; j < 8; ++j) pk[j] = src[(size_t)j * H * Wd];
        }
        *(v8bf*)&Bt[n * 32 + kg * 8] = pk;        // one b128 LDS store
      }
      if (kc + 32 < Cin)                          // global_prefetch next chunk
        __builtin_prefetch(&wtap[(size_t)(tid & 127) * Cin + kc + 32], 0, 1);
      __syncthreads();
      v16bf a = load_a_frag(Aw + wv * 16 * 32, 32, 0);
#pragma unroll
      for (int nt = 0; nt < 4; ++nt) {
        v16bf b = load_b_frag(Bt + nt * 16 * 32, 32, 0);
        acc[nt] = wmma_bf16(a, b, acc[nt]);
      }
    }
  }
#pragma unroll
  for (int v = 0; v < 8; ++v) {
    int m = mb * 128 + wv * 16 + v + 8 * g;
    float s = gamma[m] * rsqrtf(rvar[m] + 1e-5f);
    float t = (bm[m] - rmean[m]) * s + beta[m];    // y = acc*s + t, then ReLU
    size_t ob = ((size_t)(batch * Cin + m) * H + h) * Wd + w0;
#pragma unroll
    for (int nt = 0; nt < 4; ++nt) {
      float y = acc[nt][v] * s + t;
      Outp[ob + nt * 16 + ln] = fmaxf(y, 0.f);
    }
  }
}

// ---- small converters -----------------------------------------------------
__global__ void __launch_bounds__(256)
k_cvt_x(const float* __restrict__ X, __bf16* __restrict__ Cat)
{
  const size_t i = ((size_t)blockIdx.x * 256 + threadIdx.x) * 4;  // 16.7M elems
  size_t b = i >> 21;                     // c*hw = 2^21 per batch
  size_t r = i & 2097151ull;
  v4f x = *(const v4f*)&X[i];
  v4bf o;
  o[0] = (__bf16)x[0]; o[1] = (__bf16)x[1];
  o[2] = (__bf16)x[2]; o[3] = (__bf16)x[3];
  *(v4bf*)&Cat[((size_t)b * 384 + 256) * 16384 + r] = o;  // concat block 2
}

__global__ void __launch_bounds__(256)
k_cvt_wm(const float* __restrict__ Wm, __bf16* __restrict__ Wt)
{
  int f = blockIdx.x * 256 + threadIdx.x;          // 384*384*9 = 1327104
  if (f >= 384 * 384 * 9) return;
  int kw = f % 3, kh = (f / 3) % 3, ic = (f / 9) % 384, o = f / (9 * 384);
  Wt[(size_t)(kh * 3 + kw) * 384 * 384 + (size_t)o * 384 + ic] = (__bf16)Wm[f];
}

// ---------------------------------------------------------------------------
extern "C" void kernel_launch(void* const* d_in, const int* in_sizes, int n_in,
                              void* d_out, int out_size, void* d_ws, size_t ws_size,
                              hipStream_t stream)
{
  (void)in_sizes; (void)n_in; (void)out_size; (void)ws_size;
  const float* x   = (const float*)d_in[0];
  const float* xf  = (const float*)d_in[1];
  const float* xr  = (const float*)d_in[2];
  const float* Wx  = (const float*)d_in[3];
  const float* bx  = (const float*)d_in[4];
  const float* W1  = (const float*)d_in[5];
  const float* b1  = (const float*)d_in[6];
  const float* W2  = (const float*)d_in[7];
  const float* b2  = (const float*)d_in[8];
  const float* Wf  = (const float*)d_in[9];
  const float* bf_ = (const float*)d_in[10];
  const float* Wr  = (const float*)d_in[11];
  const float* br  = (const float*)d_in[12];
  const float* Wm  = (const float*)d_in[13];
  const float* bm  = (const float*)d_in[14];
  const float* gam = (const float*)d_in[15];
  const float* bet = (const float*)d_in[16];
  const float* rme = (const float*)d_in[17];
  const float* rva = (const float*)d_in[18];
  float* outp = (float*)d_out;

  const int B = 8, C = 128, H = 128, Wd = 128, HW = H * Wd;
  const size_t NC = (size_t)B * C * HW;            // 16,777,216
  __bf16* xp  = (__bf16*)d_ws;
  __bf16* x1b = xp + NC;
  __bf16* x2b = x1b + NC;
  __bf16* x2c = x2b;                               // alias (readability)
  __bf16* xfp = x2b + NC;
  __bf16* xrp = xfp + NC;
  __bf16* cat = xrp + NC;                          // 3*NC (concat buffer)
  __bf16* wt  = cat + 3 * NC;                      // 9*384*384 tap weights
  (void)x2c;

  dim3 blk(256);
  k_cvt_x <<<dim3(16384), blk, 0, stream>>>(x, cat);
  k_cvt_wm<<<dim3(5184),  blk, 0, stream>>>(Wm, wt);

  dim3 g1(HW / 64, B);
  k_gemm1x1<false><<<g1, blk, 0, stream>>>(Wx, bx,  x,  nullptr, xp,  C, HW);
  k_gemm1x1<false><<<g1, blk, 0, stream>>>(Wf, bf_, xf, nullptr, xfp, C, HW);
  k_gemm1x1<false><<<g1, blk, 0, stream>>>(Wr, br,  xr, nullptr, xrp, C, HW);
  k_gemm1x1<true> <<<g1, blk, 0, stream>>>(W1, b1,  nullptr, xp, x1b, C, HW);
  k_gemm1x1<true> <<<g1, blk, 0, stream>>>(W2, b2,  nullptr, xp, x2b, C, HW);

  dim3 g2(H, B);
  k_attn<<<g2, blk, 0, stream>>>(x1b, xfp, xp, cat, 0);
  k_attn<<<g2, blk, 0, stream>>>(x2b, xrp, xp, cat, 1);

  dim3 g3(2 * H, 3, B);
  k_conv3x3<<<g3, blk, 0, stream>>>(cat, wt, bm, gam, bet, rme, rva, outp);
}
